// HybridAE_15126874817281
// MI455X (gfx1250) — 16-bit and 8-bit matrix operand order repaired and checked by reading, compiled, not run on hardware
//
#include <hip/hip_runtime.h>
#include <cstdint>

#define N_FEATS 784
#define N_HID   256
#define N_CODE  64
#define EPSQ    1e-5f

typedef __attribute__((ext_vector_type(8)))  int      v8i;
typedef __attribute__((ext_vector_type(8)))  float    v8f;
typedef __attribute__((ext_vector_type(16))) _Float16 v16h;

#define SC_OFF   0
#define W1Q_DW   (13*16*32*8)
#define W1Q_OFF  256
#define W2Q_DW   (4*4*32*8)
#define W2Q_OFF  (W1Q_OFF + W1Q_DW*4)
#define WDQ_DW   (49*2*32*8)
#define WDQ_OFF  (W2Q_OFF + W2Q_DW*4)

__global__ void k_scales(const float* __restrict__ W1, const float* __restrict__ W2,
                         float* __restrict__ sc) {
  __shared__ float red[256];
  const int t = threadIdx.x;
  float s = 0.f;
  for (int i = t; i < N_HID * N_FEATS; i += 256) s += fabsf(W1[i]);
  red[t] = s; __syncthreads();
  for (int off = 128; off; off >>= 1) { if (t < off) red[t] += red[t + off]; __syncthreads(); }
  if (t == 0) sc[0] = fmaxf(red[0] / (float)(N_HID * N_FEATS), EPSQ);
  __syncthreads();
  s = 0.f;
  for (int i = t; i < N_CODE * N_HID; i += 256) s += fabsf(W2[i]);
  red[t] = s; __syncthreads();
  for (int off = 128; off; off >>= 1) { if (t < off) red[t] += red[t + off]; __syncthreads(); }
  if (t == 0) sc[1] = fmaxf(red[0] / (float)(N_CODE * N_HID), EPSQ);
}

__global__ void k_packW1(const float* __restrict__ W1, const float* __restrict__ sc,
                         unsigned* __restrict__ w1q) {
  int idx = blockIdx.x * 256 + threadIdx.x;
  if (idx >= W1Q_DW) return;
  int j = idx & 7, lane = (idx >> 3) & 31, nt = (idx >> 8) & 15, kt = idx >> 12;
  int half = lane >> 4, n = nt * 16 + (lane & 15);
  float inv = 1.0f / sc[0];
  int kb = kt * 64 + ((j >> 2) << 5) + (half << 4) + ((j & 3) << 2);
  unsigned d = 0;
  for (int b = 0; b < 4; ++b) {
    int k = kb + b, q = 0;
    if (k < N_FEATS) {
      q = (int)rintf(W1[n * N_FEATS + k] * inv);
      q = q < -1 ? -1 : (q > 1 ? 1 : q);
    }
    d |= ((unsigned)(q & 0xff)) << (8 * b);
  }
  w1q[idx] = d;
}

__global__ void k_packW2(const float* __restrict__ W2, const float* __restrict__ sc,
                         unsigned* __restrict__ w2q) {
  int idx = blockIdx.x * 256 + threadIdx.x;
  if (idx >= W2Q_DW) return;
  int j = idx & 7, lane = (idx >> 3) & 31, nt = (idx >> 8) & 3, kt = (idx >> 10) & 3;
  int half = lane >> 4, n = nt * 16 + (lane & 15);
  float inv = 1.0f / sc[1];
  int kb = kt * 64 + ((j >> 2) << 5) + (half << 4) + ((j & 3) << 2);
  unsigned d = 0;
  for (int b = 0; b < 4; ++b) {
    int q = (int)rintf(W2[n * N_HID + kb + b] * inv);
    q = q < -1 ? -1 : (q > 1 ? 1 : q);
    d |= ((unsigned)(q & 0xff)) << (8 * b);
  }
  w2q[idx] = d;
}

__global__ void k_packWd(const float* __restrict__ Wd, unsigned* __restrict__ wdq) {
  int idx = blockIdx.x * 256 + threadIdx.x;
  if (idx >= WDQ_DW) return;
  int j = idx & 7, lane = (idx >> 3) & 31, kt = (idx >> 8) & 1, nt = idx >> 9;
  int half = lane >> 4, n = nt * 16 + (lane & 15);
  int k = kt * 32 + (half << 4) + (j << 1);
  union { unsigned u; _Float16 f[2]; } p;
  p.f[0] = (_Float16)Wd[n * N_CODE + k];
  p.f[1] = (_Float16)Wd[n * N_CODE + k + 1];
  wdq[idx] = p.u;
}

__device__ __forceinline__ unsigned q8pack(float a, float b, float c, float d, float s) {
  int q0 = (int)rintf(a * s); q0 = q0 < -128 ? -128 : (q0 > 127 ? 127 : q0);
  int q1 = (int)rintf(b * s); q1 = q1 < -128 ? -128 : (q1 > 127 ? 127 : q1);
  int q2 = (int)rintf(c * s); q2 = q2 < -128 ? -128 : (q2 > 127 ? 127 : q2);
  int q3 = (int)rintf(d * s); q3 = q3 < -128 ? -128 : (q3 > 127 ? 127 : q3);
  return (unsigned)(q0 & 0xff) | ((unsigned)(q1 & 0xff) << 8) |
         ((unsigned)(q2 & 0xff) << 16) | ((unsigned)(q3 & 0xff) << 24);
}

__global__ __launch_bounds__(256) void k_main(
    const float* __restrict__ x,  const float* __restrict__ b1,
    const float* __restrict__ b2, const float* __restrict__ bd,
    const float* __restrict__ sc,
    const unsigned* __restrict__ w1q, const unsigned* __restrict__ w2q,
    const unsigned* __restrict__ wdq, float* __restrict__ out) {
  __shared__ float s1buf[8][16];
  __shared__ __align__(16) signed char hbuf[8][16][N_HID];
  __shared__ __align__(16) _Float16    cbuf[8][16][N_CODE];

  const int lane = threadIdx.x & 31;
  const int wv   = threadIdx.x >> 5;
  const int half = lane >> 4;
  const int m    = lane & 15;
  const long row0 = ((long)blockIdx.x * 8 + wv) * 16;
  const float scale1 = sc[0], scale2 = sc[1];

  {
    const float4* xr = (const float4*)(x + (row0 + m) * N_FEATS + half * 392);
    float mx = 0.f;
    #pragma unroll 4
    for (int i = 0; i < 98; ++i) {
      float4 v = xr[i];
      mx = fmaxf(mx, fmaxf(fmaxf(fabsf(v.x), fabsf(v.y)), fmaxf(fabsf(v.z), fabsf(v.w))));
    }
    mx = fmaxf(mx, __shfl_xor(mx, 16, 32));
    if (half == 0) s1buf[wv][m] = 127.0f / fmaxf(mx, EPSQ);
  }
  __syncthreads();

  const float s_my = s1buf[wv][m];
  float srow[8];
  #pragma unroll
  for (int r = 0; r < 8; ++r) srow[r] = s1buf[wv][r + 8 * half];

  const v8i zi = {0, 0, 0, 0, 0, 0, 0, 0};
  v8i acc[16];
  #pragma unroll
  for (int nt = 0; nt < 16; ++nt) acc[nt] = zi;
  const float* xrow = x + (row0 + m) * N_FEATS;
  for (int kt = 0; kt < 13; ++kt) {
    union { v8i v; unsigned u[8]; } A;
    #pragma unroll
    for (int j = 0; j < 8; ++j) {
      int kb = kt * 64 + ((j >> 2) << 5) + (half << 4) + ((j & 3) << 2);
      unsigned d = 0;
      if (kb < N_FEATS) {
        float4 v = *(const float4*)(xrow + kb);
        d = q8pack(v.x, v.y, v.z, v.w, s_my);
      }
      A.u[j] = d;
    }
    const v8i* Bp = (const v8i*)w1q + (size_t)(kt * 16) * 32;
    #pragma unroll
    for (int nt = 0; nt < 16; ++nt) {
      v8i B = Bp[nt * 32 + lane];
      acc[nt] = __builtin_amdgcn_wmma_i32_16x16x64_iu8(true, A.v, true, B, acc[nt], false, false);
    }
  }

  float h[16][8], dq1[8], mx2[8];
  #pragma unroll
  for (int r = 0; r < 8; ++r) { dq1[r] = scale1 / srow[r]; mx2[r] = 0.f; }
  #pragma unroll
  for (int nt = 0; nt < 16; ++nt) {
    float bn = b1[nt * 16 + m];
    #pragma unroll
    for (int r = 0; r < 8; ++r) {
      float v = (float)acc[nt][r] * dq1[r] + bn;
      v = v / (1.0f + __expf(-v));
      h[nt][r] = v;
      mx2[r] = fmaxf(mx2[r], fabsf(v));
    }
  }
  float s2row[8];
  #pragma unroll
  for (int r = 0; r < 8; ++r) {
    float v = mx2[r];
    v = fmaxf(v, __shfl_xor(v, 1, 32));
    v = fmaxf(v, __shfl_xor(v, 2, 32));
    v = fmaxf(v, __shfl_xor(v, 4, 32));
    v = fmaxf(v, __shfl_xor(v, 8, 32));
    s2row[r] = 127.0f / fmaxf(v, EPSQ);
  }
  #pragma unroll
  for (int nt = 0; nt < 16; ++nt)
    #pragma unroll
    for (int r = 0; r < 8; ++r) {
      int q = (int)rintf(h[nt][r] * s2row[r]);
      q = q < -128 ? -128 : (q > 127 ? 127 : q);
      hbuf[wv][r + 8 * half][nt * 16 + m] = (signed char)q;
    }
  __syncthreads();

  v8i acc2[4];
  #pragma unroll
  for (int nt = 0; nt < 4; ++nt) acc2[nt] = zi;
  const signed char* hrow = &hbuf[wv][m][0];
  #pragma unroll
  for (int kt = 0; kt < 4; ++kt) {
    union { v8i v; unsigned u[8]; } A;
    #pragma unroll
    for (int j = 0; j < 8; ++j) {
      int kb = kt * 64 + ((j >> 2) << 5) + (half << 4) + ((j & 3) << 2);
      A.u[j] = *(const unsigned*)(hrow + kb);
    }
    const v8i* Bp = (const v8i*)w2q + (size_t)(kt * 4) * 32;
    #pragma unroll
    for (int nt = 0; nt < 4; ++nt) {
      v8i B = Bp[nt * 32 + lane];
      acc2[nt] = __builtin_amdgcn_wmma_i32_16x16x64_iu8(true, A.v, true, B, acc2[nt], false, false);
    }
  }

  float dq2[8];
  #pragma unroll
  for (int r = 0; r < 8; ++r) dq2[r] = scale2 / s2row[r];
  #pragma unroll
  for (int nt = 0; nt < 4; ++nt) {
    float bn = b2[nt * 16 + m];
    #pragma unroll
    for (int r = 0; r < 8; ++r) {
      float c = (float)acc2[nt][r] * dq2[r] + bn;
      cbuf[wv][r + 8 * half][nt * 16 + m] = (_Float16)c;
    }
  }
  __syncthreads();

  union { v16h v; unsigned u[8]; } A3[2];
  const _Float16* crow = &cbuf[wv][m][0];
  #pragma unroll
  for (int kt = 0; kt < 2; ++kt)
    #pragma unroll
    for (int j = 0; j < 8; ++j) {
      int kb = kt * 32 + (half << 4) + (j << 1);
      A3[kt].u[j] = *(const unsigned*)(crow + kb);
    }
  const v16h* Bw = (const v16h*)wdq;
  float* orow = out + row0 * N_FEATS;
  const v8f zf = {0.f, 0.f, 0.f, 0.f, 0.f, 0.f, 0.f, 0.f};
  for (int nt = 0; nt < 49; ++nt) {
    v16h B0 = Bw[(nt * 2 + 0) * 32 + lane];
    v16h B1 = Bw[(nt * 2 + 1) * 32 + lane];
    v8f c = zf;
    c = __builtin_amdgcn_wmma_f32_16x16x32_f16(false, A3[0].v, false, B0, (short)0, c, false, false);
    c = __builtin_amdgcn_wmma_f32_16x16x32_f16(false, A3[1].v, false, B1, (short)0, c, false, false);
    float bdv = bd[nt * 16 + m];
    #pragma unroll
    for (int r = 0; r < 8; ++r)
      orow[(long)(r + 8 * half) * N_FEATS + nt * 16 + m] = c[r] + bdv;
  }
}

extern "C" void kernel_launch(void* const* d_in, const int* in_sizes, int n_in,
                              void* d_out, int out_size, void* d_ws, size_t ws_size,
                              hipStream_t stream) {
  const float* x  = (const float*)d_in[0];
  const float* W1 = (const float*)d_in[1];
  const float* b1 = (const float*)d_in[2];
  const float* W2 = (const float*)d_in[3];
  const float* b2 = (const float*)d_in[4];
  const float* Wd = (const float*)d_in[5];
  const float* bd = (const float*)d_in[6];
  float* out = (float*)d_out;
  char* ws = (char*)d_ws;
  float*    sc  = (float*)(ws + SC_OFF);
  unsigned* w1q = (unsigned*)(ws + W1Q_OFF);
  unsigned* w2q = (unsigned*)(ws + W2Q_OFF);
  unsigned* wdq = (unsigned*)(ws + WDQ_OFF);

  const int rows = in_sizes[0] / N_FEATS;
  k_scales<<<1, 256, 0, stream>>>(W1, W2, sc);
  k_packW1<<<(W1Q_DW + 255) / 256, 256, 0, stream>>>(W1, sc, w1q);
  k_packW2<<<(W2Q_DW + 255) / 256, 256, 0, stream>>>(W2, sc, w2q);
  k_packWd<<<(WDQ_DW + 255) / 256, 256, 0, stream>>>(Wd, wdq);
  const int blocks = rows / (16 * 8);
  k_main<<<blocks, 256, 0, stream>>>(x, b1, b2, bd, sc, w1q, w2q, wdq, out);
}
